// InfoNCE_contrastiveLoss_3092376453680
// MI455X (gfx1250) — matrix operand order repaired and checked by reading, not compiled, not run on hardware
//
#include <hip/hip_runtime.h>
#include <hip/hip_fp16.h>

#define NROWS 8192
#define DIM   1024
#define INV_T 10.0f

typedef __attribute__((ext_vector_type(16))) _Float16 v16h;
typedef __attribute__((ext_vector_type(8)))  _Float16 v8h;
typedef __attribute__((ext_vector_type(4)))  _Float16 v4h;
typedef __attribute__((ext_vector_type(8)))  float    v8f;

// ---------------------------------------------------------------- zero
__global__ __launch_bounds__(256) void zero_kernel(float* __restrict__ p, int n) {
  int i = blockIdx.x * 256 + threadIdx.x;
  if (i < n) p[i] = 0.0f;
}

// ------------------------------------------------- row L2-normalize -> f16
// grid: (NROWS, 2); block: 256 threads, 4 floats/thread (row of 1024)
__global__ __launch_bounds__(256) void normalize_kernel(
    const float* __restrict__ im, const float* __restrict__ cap,
    _Float16* __restrict__ imh, _Float16* __restrict__ caph) {
  const int row = blockIdx.x;
  const float* __restrict__ src = blockIdx.y ? cap : im;
  _Float16* __restrict__ dst    = blockIdx.y ? caph : imh;

  const float4 v = ((const float4*)(src + (size_t)row * DIM))[threadIdx.x];
  float ss = v.x * v.x + v.y * v.y + v.z * v.z + v.w * v.w;
  #pragma unroll
  for (int off = 1; off < 32; off <<= 1) ss += __shfl_xor(ss, off, 32);

  __shared__ float wsum[8];
  __shared__ float sscale;
  const int lane = threadIdx.x & 31, wv = threadIdx.x >> 5;
  if (lane == 0) wsum[wv] = ss;
  __syncthreads();
  if (threadIdx.x == 0) {
    float t = 0.0f;
    #pragma unroll
    for (int i = 0; i < 8; ++i) t += wsum[i];
    sscale = rsqrtf(t);
  }
  __syncthreads();
  const float sc = sscale;

  v4h o;
  o[0] = (_Float16)(v.x * sc);
  o[1] = (_Float16)(v.y * sc);
  o[2] = (_Float16)(v.z * sc);
  o[3] = (_Float16)(v.w * sc);
  *(v4h*)(dst + (size_t)row * DIM + (size_t)threadIdx.x * 4) = o;
}

// -------------------------------------------- fused WMMA GEMM + exp + LSE sums
// WG tile 256(M) x 128(N); 8 waves as 4x2, each wave 64x64 (4x4 WMMA frags).
// grid: (8192/128, 8192/256) = (64, 32)
__global__ __launch_bounds__(256) void gemm_lse_kernel(
    const _Float16* __restrict__ imh, const _Float16* __restrict__ caph,
    float* __restrict__ rowsum, float* __restrict__ colsum,
    float* __restrict__ diag) {
  const int lane = threadIdx.x & 31;
  const int wv   = threadIdx.x >> 5;
  const int half = lane >> 4;   // 0: lanes 0-15, 1: lanes 16-31
  const int l15  = lane & 15;

  const int tileM = blockIdx.y * 256;
  const int tileN = blockIdx.x * 128;
  const int waveM = tileM + (wv >> 1) * 64;
  const int waveN = tileN + (wv & 1) * 64;

  v8f acc[4][4];
  {
    v8f z = {0.f, 0.f, 0.f, 0.f, 0.f, 0.f, 0.f, 0.f};
    #pragma unroll
    for (int mi = 0; mi < 4; ++mi)
      #pragma unroll
      for (int ni = 0; ni < 4; ++ni) acc[mi][ni] = z;
  }

  // A fragment (16x32 f16): lane half 0 holds k+0..15, half 1 holds k+16..31
  const _Float16* aPtr[4];
  // B fragment (32x16 f16), B = cap^T: lane l15 = column n -> cap row n,
  // half 0 holds k+0..15, half 1 holds k+16..31 (contiguous 32B)
  const _Float16* bPtr[4];
  #pragma unroll
  for (int mi = 0; mi < 4; ++mi)
    aPtr[mi] = imh + (size_t)(waveM + 16 * mi + l15) * DIM + 16 * half;
  #pragma unroll
  for (int ni = 0; ni < 4; ++ni)
    bPtr[ni] = caph + (size_t)(waveN + 16 * ni + l15) * DIM + 16 * half;

  for (int k0 = 0; k0 < DIM; k0 += 32) {
    v16h a[4], b[4];
    #pragma unroll
    for (int mi = 0; mi < 4; ++mi) {
      const v8h* p = (const v8h*)(aPtr[mi] + k0);
      v8h lo = p[0];        // k+off+0..7
      v8h hi = p[1];        // k+off+8..15
      a[mi] = __builtin_shufflevector(lo, hi,
                0, 1, 2, 3, 4, 5, 6, 7, 8, 9, 10, 11, 12, 13, 14, 15);
    }
    #pragma unroll
    for (int ni = 0; ni < 4; ++ni)
      b[ni] = *(const v16h*)(bPtr[ni] + k0);

    #pragma unroll
    for (int mi = 0; mi < 4; ++mi)
      #pragma unroll
      for (int ni = 0; ni < 4; ++ni)
        acc[mi][ni] = __builtin_amdgcn_wmma_f32_16x16x32_f16(
            false, a[mi], false, b[ni], (short)0, acc[mi][ni], false, false);
  }

  // C/D layout: element (m,n) of a 16x16 tile lives at VGPR r=m%8,
  // lane = (n) + 16*(m/8).  gm = waveM+16*mi+8*half+r, gn = waveN+16*ni+l15.

  // ---- diagonal (only tiles straddling it; each (i,i) has a unique writer)
  if (tileM < tileN + 128 && tileN < tileM + 256) {
    #pragma unroll
    for (int mi = 0; mi < 4; ++mi)
      #pragma unroll
      for (int ni = 0; ni < 4; ++ni) {
        const int gm0 = waveM + 16 * mi + 8 * half;
        const int gn  = waveN + 16 * ni + l15;
        #pragma unroll
        for (int r = 0; r < 8; ++r)
          if (gm0 + r == gn) diag[gn] = acc[mi][ni][r] * INV_T;
      }
  }

  // ---- exp + row/column partial sums (|S|<=10 so no max pass needed)
  float rAcc[4][8];
  #pragma unroll
  for (int mi = 0; mi < 4; ++mi)
    #pragma unroll
    for (int r = 0; r < 8; ++r) rAcc[mi][r] = 0.0f;

  #pragma unroll
  for (int ni = 0; ni < 4; ++ni) {
    float csum = 0.0f;
    #pragma unroll
    for (int mi = 0; mi < 4; ++mi)
      #pragma unroll
      for (int r = 0; r < 8; ++r) {
        float e = __expf(acc[mi][ni][r] * INV_T);
        rAcc[mi][r] += e;   // this lane's column contribution to row gm
        csum += e;          // rows {16*mi + 8*half + r} of column gn
      }
    csum += __shfl_xor(csum, 16, 32);          // combine both lane halves
    if (half == 0) atomicAdd(&colsum[waveN + 16 * ni + l15], csum);
  }

  #pragma unroll
  for (int mi = 0; mi < 4; ++mi)
    #pragma unroll
    for (int r = 0; r < 8; ++r) {
      float v = rAcc[mi][r];
      v += __shfl_xor(v, 1, 32);
      v += __shfl_xor(v, 2, 32);
      v += __shfl_xor(v, 4, 32);
      v += __shfl_xor(v, 8, 32);               // sum across 16 columns
      if (l15 == 0)
        atomicAdd(&rowsum[waveM + 16 * mi + 8 * half + r], v);
    }
}

// ---------------------------------------------------------------- finalize
__global__ __launch_bounds__(256) void finalize_kernel(
    const float* __restrict__ rowsum, const float* __restrict__ colsum,
    const float* __restrict__ diag, float* __restrict__ out) {
  float acc = 0.0f;
  for (int i = threadIdx.x; i < NROWS; i += 256)
    acc += 0.5f * (logf(rowsum[i]) + logf(colsum[i])) - diag[i];
  #pragma unroll
  for (int off = 1; off < 32; off <<= 1) acc += __shfl_xor(acc, off, 32);
  __shared__ float wsum[8];
  const int lane = threadIdx.x & 31, wv = threadIdx.x >> 5;
  if (lane == 0) wsum[wv] = acc;
  __syncthreads();
  if (threadIdx.x == 0) {
    float t = 0.0f;
    #pragma unroll
    for (int i = 0; i < 8; ++i) t += wsum[i];
    out[0] = t / (float)NROWS;
  }
}

// ---------------------------------------------------------------- launch
extern "C" void kernel_launch(void* const* d_in, const int* in_sizes, int n_in,
                              void* d_out, int out_size, void* d_ws, size_t ws_size,
                              hipStream_t stream) {
  const float* im  = (const float*)d_in[0];
  const float* cap = (const float*)d_in[1];
  float* out = (float*)d_out;

  // workspace layout: imh[N*D] f16 | caph[N*D] f16 | rowsum[N] | colsum[N] | diag[N]
  _Float16* imh  = (_Float16*)d_ws;
  _Float16* caph = imh + (size_t)NROWS * DIM;
  float* rowsum  = (float*)(caph + (size_t)NROWS * DIM);
  float* colsum  = rowsum + NROWS;
  float* diag    = colsum + NROWS;

  zero_kernel<<<(2 * NROWS + 255) / 256, 256, 0, stream>>>(rowsum, 2 * NROWS);
  normalize_kernel<<<dim3(NROWS, 2), 256, 0, stream>>>(im, cap, imh, caph);
  gemm_lse_kernel<<<dim3(NROWS / 128, NROWS / 256), 256, 0, stream>>>(
      imh, caph, rowsum, colsum, diag);
  finalize_kernel<<<1, 256, 0, stream>>>(rowsum, colsum, diag, out);
}
